// WeatherForecastingModel_45440753991824
// MI455X (gfx1250) — matrix operand order repaired and checked by reading, not compiled, not run on hardware
//
#include <hip/hip_runtime.h>

// ---------------- WMMA types ----------------
typedef __attribute__((ext_vector_type(16))) __bf16 v16bf;
typedef __attribute__((ext_vector_type(8)))  float  v8f;

#define HH    128        // hidden size
#define BB    8          // batch
#define TT    168        // seq len / forecast len
#define GG    512        // 4*H gate width
#define KCH   4          // K chunks of 32 (K=128)
#define WSLOT 2048       // bf16 elements per h0 handoff slot (16x128)

// bf16 round-to-nearest-even
__device__ __forceinline__ unsigned short f2bf(float x) {
  union { float f; unsigned u; } v; v.f = x;
  unsigned u = v.u;
  unsigned r = (u + 0x7FFFu + ((u >> 16) & 1u)) >> 16;
  return (unsigned short)r;
}
__device__ __forceinline__ float sigf(float x) { return 1.0f / (1.0f + __expf(-x)); }

// B fragment: 32 contiguous bytes per lane (prepacked stream, LDS or global)
__device__ __forceinline__ v16bf load_b_frag(const unsigned short* p) {
  v16bf r;
  uint4* rp = reinterpret_cast<uint4*>(&r);
  const uint4* sp = reinterpret_cast<const uint4*>(p);
  rp[0] = sp[0]; rp[1] = sp[1];
  return r;
}

// A fragment from a row-major 16x128 bf16 buffer (LDS stage or global h0 slot).
// A layout (matches B pack): lanes 0-15: M=lane, K=kc*32+{0..15}; lanes 16-31: M=lane-16, K=kc*32+{16..31}
__device__ __forceinline__ v16bf load_a_frag(const unsigned short* stage, int lane, int kc) {
  int row  = lane & 15;
  int koff = (lane < 16) ? 0 : 16;
  const unsigned short* base = stage + row * HH + kc * 32 + koff;
  v16bf r;
  uint4* rp = reinterpret_cast<uint4*>(&r);
  rp[0] = *reinterpret_cast<const uint4*>(base);
  rp[1] = *reinterpret_cast<const uint4*>(base + 8);
  return r;
}

// ---------------- cross-WGP split-barrier flags ----------------
__device__ __forceinline__ void wait_ge(int* p, int target, int tid) {
  if (tid == 0) {
    while (__hip_atomic_load(p, __ATOMIC_ACQUIRE, __HIP_MEMORY_SCOPE_AGENT) < target)
      __builtin_amdgcn_s_sleep(1);
  }
  __syncthreads();
}
__device__ __forceinline__ void post(int* p, int val, int tid) {
  __threadfence();        // each thread's global writes visible at device scope
  __syncthreads();        // all threads done
  if (tid == 0)
    __hip_atomic_store(p, val, __ATOMIC_RELEASE, __HIP_MEMORY_SCOPE_AGENT);
}

// One LSTM layer step. 32 waves: wave w owns gate columns [w*16, w*16+16).
// Weights come from this WGP's LDS (ds_load_b128); A from LDS stage or global h0 slot.
__device__ __forceinline__ void lstm_cell(
    const unsigned short* A_in,          // layer input, row-major 16x128 bf16
    const unsigned short* w_ih_lds,      // LDS packed [32][4][512]
    const unsigned short* w_hh_lds,      // LDS packed [32][4][512]
    unsigned short*       hbf,           // LDS 16x128 bf16 h-state (rows 8-15 stay 0)
    float*                cst,           // LDS 8x128 f32 c-state
    const float*          biasl,         // LDS 512 f32 (b_ih + b_hh)
    float*                gates,         // LDS 16x512 f32 exchange
    unsigned short*       hout_bf,       // optional global bf16 h-out slot (producer)
    float*                gout,          // optional global f32 h-out (consumer -> buf)
    int tid)
{
  const int lane = tid & 31;
  const int w    = tid >> 5;

  // launder the weight base index so the per-step ds_load_b128s cannot be
  // hoisted out of the sequential time loop (previous round: scratch spills)
  int wbase = w * (KCH * 512);
  asm volatile("" : "+v"(wbase));

  v8f acc = {0.f,0.f,0.f,0.f,0.f,0.f,0.f,0.f};
  const unsigned short* bih = w_ih_lds + wbase;
  const unsigned short* bhh = w_hh_lds + wbase;

#pragma unroll
  for (int kc = 0; kc < KCH; ++kc) {
    v16bf a = load_a_frag(A_in, lane, kc);
    v16bf b = load_b_frag(bih + kc * 512 + lane * 16);
    acc = __builtin_amdgcn_wmma_f32_16x16x32_bf16(false, a, false, b, (short)0, acc, false, false);
  }
#pragma unroll
  for (int kc = 0; kc < KCH; ++kc) {
    v16bf a = load_a_frag(hbf, lane, kc);
    v16bf b = load_b_frag(bhh + kc * 512 + lane * 16);
    acc = __builtin_amdgcn_wmma_f32_16x16x32_bf16(false, a, false, b, (short)0, acc, false, false);
  }

  // C/D layout: VGPR r, lanes 0-15 -> (M=r, N=lane); lanes 16-31 -> (M=8+r, N=lane-16)
  const int mbase = (lane < 16) ? 0 : 8;
  const int col   = w * 16 + (lane & 15);
#pragma unroll
  for (int r = 0; r < 8; ++r)
    gates[(mbase + r) * GG + col] = acc[r];
  __syncthreads();

  // elementwise i,f,g,o -> c,h : 1024 threads, one live element each (rows 0-7)
  {
    const int e = tid;
    const int m = e >> 7, j = e & 127;
    float gi = gates[m * GG + j]        + biasl[j];
    float gf = gates[m * GG + HH + j]   + biasl[HH + j];
    float gg = gates[m * GG + 2*HH + j] + biasl[2*HH + j];
    float go = gates[m * GG + 3*HH + j] + biasl[3*HH + j];
    float cp = cst[e];
    float cn = sigf(gf) * cp + sigf(gi) * tanhf(gg);
    float hn = sigf(go) * tanhf(cn);
    unsigned short hb = f2bf(hn);
    cst[e] = cn;
    hbf[e] = hb;
    if (hout_bf) hout_bf[e] = hb;
    if (gout)    gout[e]    = hn;
  }
  __syncthreads();
}

// -------- prep: pack fp32 weights into bf16 WMMA B-fragment stream; zero flags --------
// wpack block layout: (((L*2+mat)*32 + ntile)*4 + kchunk)*512 bf16;
// within block: lane*16+e ; n = ntile*16+(lane&15) ; k = kchunk*32+(lane>>4)*16+e
__global__ void wf_prep_pack(const float* __restrict__ W_ih, const float* __restrict__ W_hh,
                             const float* __restrict__ W_fc,
                             unsigned short* __restrict__ wpack,
                             unsigned short* __restrict__ fcpack,
                             int* flags)
{
  int g = blockIdx.x * blockDim.x + threadIdx.x;
  if (g < 2) flags[g] = 0;
  if (g < 262144) {
    int blk = g >> 9, in = g & 511;
    int lane = in >> 4, e = in & 15;
    int kc = blk & 3, nt = (blk >> 2) & 31, mat = (blk >> 7) & 1, l = blk >> 8;
    int n = nt * 16 + (lane & 15);
    int k = kc * 32 + (lane >> 4) * 16 + e;
    const float* W = mat ? W_hh : W_ih;
    wpack[g] = f2bf(W[(l * GG + n) * HH + k]);
  } else if (g < 262144 + 16384) {
    int g2 = g - 262144;
    int blk = g2 >> 9, in = g2 & 511;
    int lane = in >> 4, e = in & 15;
    int kc = blk & 3, nt = blk >> 2;
    int n = nt * 16 + (lane & 15);
    int k = kc * 32 + (lane >> 4) * 16 + e;
    fcpack[g2] = f2bf(W_fc[n * HH + k]);
  }
}

// -------- persistent pipelined kernel: block 0 = layer 0, block 1 = layer 1 --------
__global__ __launch_bounds__(1024) void wf_pipe(
    const float* __restrict__ x,        // [8][168][128]
    const float* __restrict__ b_ih,     // [2][512]
    const float* __restrict__ b_hh,     // [2][512]
    const float* __restrict__ b_fc,     // [128]
    const unsigned short* __restrict__ wpack,   // packed bf16 weights (both layers)
    const unsigned short* __restrict__ fcpack,  // packed bf16 W_fc
    float*          buf,                // ws: [169][8][128] replay buffer
    unsigned short* h0buf,              // ws: [2][16][128] bf16 h0 handoff
    int*            flags,              // ws: {prog0, prog1}
    float* __restrict__ out)            // [8][168*128]
{
  __shared__ unsigned short wlds[131072];      // 256 KB: this layer's Wih|Whh packed
  __shared__ float          gates[16 * GG];    // 32 KB
  __shared__ unsigned short stage_in[16 * HH]; // 4 KB (block 0 input staging)
  __shared__ unsigned short hbf[16 * HH];      // 4 KB
  __shared__ float          cst[BB * HH];      // 4 KB
  __shared__ float          biasl[GG];         // 2 KB

  const int tid   = threadIdx.x;
  const int layer = blockIdx.x;
  int* prog0 = flags;
  int* prog1 = flags + 1;

  // load this layer's 256 KB of packed weights into LDS (once)
  {
    const uint4* src = (const uint4*)(wpack + (size_t)layer * 131072);
    uint4* dst = (uint4*)wlds;
    for (int i = tid; i < 16384; i += 1024) dst[i] = src[i];
  }
  if (tid < GG) biasl[tid] = b_ih[layer * GG + tid] + b_hh[layer * GG + tid];
  cst[tid] = 0.f;
  for (int e = tid; e < 16 * HH; e += 1024) hbf[e] = 0;
  if (layer == 0)   // rows 8-15 of both handoff slots are constant zero
    for (int e = tid; e < 2048; e += 1024)
      h0buf[(e >> 10) * WSLOT + 1024 + (e & 1023)] = 0;
  __syncthreads();

  const unsigned short* wih = wlds;            // [32][4][512]
  const unsigned short* whh = wlds + 65536;

  if (layer == 0) {
    // ---------------- producer: layer 0 ----------------
    int s = 0;
    for (int t = 0; t < TT; ++t, ++s) {               // encoder
      wait_ge(prog1, s - 1, tid);                     // double-buffer throttle
      for (int e = tid; e < 16 * HH; e += 1024) {
        int m = e >> 7, j = e & 127;
        stage_in[e] = (m < BB) ? f2bf(x[(m * TT + t) * HH + j]) : (unsigned short)0;
      }
      __syncthreads();
      lstm_cell(stage_in, wih, whh, hbf, cst, biasl, gates,
                h0buf + (s & 1) * WSLOT, nullptr, tid);
      post(prog0, s + 1, tid);
    }
    for (int k = 0; k < TT; ++k) {                    // decoder replays
      wait_ge(prog1, s, tid);                         // replay k-1 + forecast drained
      for (int t = 0; t <= k; ++t, ++s) {
        wait_ge(prog1, s - 1, tid);
        for (int e = tid; e < 16 * HH; e += 1024) {
          int m = e >> 7, j = e & 127;
          stage_in[e] = (m < BB) ? f2bf(buf[t * (BB * HH) + m * HH + j]) : (unsigned short)0;
        }
        __syncthreads();
        lstm_cell(stage_in, wih, whh, hbf, cst, biasl, gates,
                  h0buf + (s & 1) * WSLOT, nullptr, tid);
        post(prog0, s + 1, tid);
      }
    }
  } else {
    // ---------------- consumer: layer 1 + forecasts ----------------
    const int lane = tid & 31, wv = tid >> 5;
    int s = 0;
    for (int t = 0; t < TT; ++t, ++s) {               // encoder
      wait_ge(prog0, s + 1, tid);
      lstm_cell(h0buf + (s & 1) * WSLOT, wih, whh, hbf, cst, biasl, gates,
                nullptr, (t == TT - 1) ? buf : nullptr, tid);  // last_out -> buf[0]
      post(prog1, s + 1, tid);
    }
    for (int k = 0; k < TT; ++k) {                    // decoder replays
      for (int t = 0; t <= k; ++t, ++s) {
        wait_ge(prog0, s + 1, tid);
        lstm_cell(h0buf + (s & 1) * WSLOT, wih, whh, hbf, cst, biasl, gates,
                  nullptr, buf + t * (BB * HH), tid); // buf[t] = outs[t]
        if (t == k) {
          // forecast = outs[k] @ W_fc^T + b_fc  (waves 0..7, N = 128)
          if (wv < 8) {
            int fb = wv * (KCH * 512);
            asm volatile("" : "+v"(fb));              // keep loads in-loop
            v8f acc = {0.f,0.f,0.f,0.f,0.f,0.f,0.f,0.f};
            const unsigned short* bfc = fcpack + fb;
#pragma unroll
            for (int kc = 0; kc < KCH; ++kc) {
              v16bf a = load_a_frag(hbf, lane, kc);
              v16bf b = load_b_frag(bfc + kc * 512 + lane * 16);
              acc = __builtin_amdgcn_wmma_f32_16x16x32_bf16(false, a, false, b, (short)0, acc, false, false);
            }
            const int mb  = (lane < 16) ? 0 : 8;
            const int col = wv * 16 + (lane & 15);
#pragma unroll
            for (int r = 0; r < 8; ++r) {
              int m = mb + r;
              if (m < BB) {
                float v = acc[r] + b_fc[col];
                out[m * (TT * HH) + k * HH + col] = v;        // output[:, k*H+col]
                buf[(k + 1) * (BB * HH) + m * HH + col] = v;  // buf[k+1] = forecast
              }
            }
          }
        }
        post(prog1, s + 1, tid);
      }
    }
  }
}

extern "C" void kernel_launch(void* const* d_in, const int* in_sizes, int n_in,
                              void* d_out, int out_size, void* d_ws, size_t ws_size,
                              hipStream_t stream) {
  (void)in_sizes; (void)n_in; (void)out_size; (void)ws_size;
  const float* x    = (const float*)d_in[0];
  const float* W_ih = (const float*)d_in[1];
  const float* W_hh = (const float*)d_in[2];
  const float* b_ih = (const float*)d_in[3];
  const float* b_hh = (const float*)d_in[4];
  const float* W_fc = (const float*)d_in[5];
  const float* b_fc = (const float*)d_in[6];
  float*       out  = (float*)d_out;

  // ws layout: wpack 512K | fcpack 32K | buf 169*8*128*4 | h0buf 2*2048*2 | flags
  unsigned short* wpack  = (unsigned short*)d_ws;
  unsigned short* fcpack = wpack + 262144;
  float*          buf    = (float*)((char*)d_ws + (size_t)(262144 + 16384) * 2);
  unsigned short* h0buf  = (unsigned short*)((char*)buf + (size_t)169 * BB * HH * 4);
  int*            flags  = (int*)((char*)h0buf + (size_t)2 * WSLOT * 2);

  wf_prep_pack<<<dim3(1088), dim3(256), 0, stream>>>(W_ih, W_hh, W_fc, wpack, fcpack, flags);
  wf_pipe<<<dim3(2), dim3(1024), 0, stream>>>(x, b_ih, b_hh, b_fc,
                                              wpack, fcpack, buf, h0buf, flags, out);
}
